// LRU_87282325389837
// MI455X (gfx1250) — compile-verified, matrix operand order checked by reading, not run on hardware
//
#include <hip/hip_runtime.h>
#include <stdint.h>

// ---------------------------------------------------------------------------
// LRU layer for MI455X (gfx1250, wave32, WMMA), bandwidth-optimized.
//   B=8, T=4096, D_IN=1024, N=512, D_OUT=1024
//   [cvt]    xbf = bf16(x)
//   [pack]   lam, lam^64 ; W1 = gamma*[B_re;B_im]; W2 = [C_re;-C_im;D] tiles
//   [gemm1]  Bu planes (bf16) = xbf @ W1   (block 16 waves, wave tile 64x64)
//   [scan]   3-pass chunked scan (64 x 64), f32 math on bf16 planes
//   [gemm2]  y = [st_re|st_im|xbf] @ W2    (block 16 waves, wave tile 64x64)
// ---------------------------------------------------------------------------

typedef __attribute__((ext_vector_type(16))) __bf16    v16bf;
typedef __attribute__((ext_vector_type(8)))  float     v8f;
typedef __attribute__((ext_vector_type(8)))  uint32_t  v8u;

#define B_SZ   8
#define T_LEN  4096
#define D_IN   1024
#define D_OUT  1024
#define N_ST   512
#define M_TOT  (B_SZ * T_LEN)          // 32768 rows
#define CHUNK  64
#define NCHNK  (T_LEN / CHUNK)         // 64
#define NT_N   64                      // 1024 cols / 16
#define MW     64                      // rows per wave tile (4 fragments)

// ---- f32 <-> bf16 helpers -------------------------------------------------
__device__ __forceinline__ unsigned short f2bf_bits(float f) {
    union { float f; uint32_t u; } v; v.f = f;
    uint32_t r = v.u + 0x7FFFu + ((v.u >> 16) & 1u);
    return (unsigned short)(r >> 16);
}
__device__ __forceinline__ float bf2f(unsigned short h) {
    union { uint32_t u; float f; } v; v.u = ((uint32_t)h) << 16;
    return v.f;
}
__device__ __forceinline__ uint32_t pack2bf(float lo, float hi) {
    return (uint32_t)f2bf_bits(lo) | ((uint32_t)f2bf_bits(hi) << 16);
}

// ---- WMMA wrapper ---------------------------------------------------------
__device__ __forceinline__ v8f wmma_bf16(v16bf a, v16bf b, v8f c) {
    return __builtin_amdgcn_wmma_f32_16x16x32_bf16(
        false, a, false, b, (short)0, c, false, false);
}

// ---- A fragment from LDS panel (bf16, row-major, rowStride = KTOT) --------
// Lane l: row = mrow + l%16; K = c..c+7 and c+16..c+23, c = k0+(l/16)*8.
__device__ __forceinline__ v16bf lds_a_frag(const unsigned short* panel,
                                            int KTOT, int mrow, int k0,
                                            int lane) {
    int m = mrow + (lane & 15);
    int c = k0 + ((lane >> 4) << 3);
    const uint4* p0 = (const uint4*)(panel + m * KTOT + c);
    const uint4* p1 = (const uint4*)(panel + m * KTOT + c + 16);
    uint4 a = *p0, b = *p1;
    v8u u;
    u[0] = a.x; u[1] = a.y; u[2] = a.z; u[3] = a.w;
    u[4] = b.x; u[5] = b.y; u[6] = b.z; u[7] = b.w;
    return __builtin_bit_cast(v16bf, u);
}

// ---- B fragment: pre-swizzled tiles, contiguous 32B per lane --------------
__device__ __forceinline__ v16bf load_b_frag(const uint32_t* __restrict__ wp,
                                             int kt, int nt, int lane) {
    const uint32_t* p = wp + ((((size_t)kt * NT_N + nt) * 32) + lane) * 8;
    v8u u = *(const v8u*)p;
    return __builtin_bit_cast(v16bf, u);
}

// ===========================================================================
// x (f32) -> xbf (bf16), 8 elements/thread
// ===========================================================================
__global__ void k_cvt_x(const float* __restrict__ src,
                        unsigned short* __restrict__ dst) {
    size_t i = ((size_t)blockIdx.x * blockDim.x + threadIdx.x) * 8;
    float4 a = *(const float4*)(src + i);
    float4 b = *(const float4*)(src + i + 4);
    uint4 o;
    o.x = pack2bf(a.x, a.y); o.y = pack2bf(a.z, a.w);
    o.z = pack2bf(b.x, b.y); o.w = pack2bf(b.z, b.w);
    *(uint4*)(dst + i) = o;
}

// ===========================================================================
// Prep: lambda and lambda^64
// ===========================================================================
__global__ void k_lam(const float* __restrict__ nu_log,
                      const float* __restrict__ theta_log,
                      float* __restrict__ lam_re, float* __restrict__ lam_im,
                      float* __restrict__ lamL_re, float* __restrict__ lamL_im) {
    int n = blockIdx.x * blockDim.x + threadIdx.x;
    if (n >= N_ST) return;
    float nu  = __expf(nu_log[n]);
    float th  = __expf(theta_log[n]);
    float mod = __expf(-nu);
    float lr = mod * __cosf(th);
    float li = mod * __sinf(th);
    lam_re[n] = lr; lam_im[n] = li;
    float r = lr, i = li;
    #pragma unroll
    for (int s = 0; s < 6; ++s) {       // lam^(2^6) = lam^64
        float nr = r * r - i * i;
        float ni = 2.0f * r * i;
        r = nr; i = ni;
    }
    lamL_re[n] = r; lamL_im[n] = i;
}

// ===========================================================================
// Weight packing into fragment-order bf16 tiles
// ===========================================================================
__device__ __forceinline__ void unpack_idx(int d, int& k, int& n) {
    int j = d & 7, l = (d >> 3) & 31, tile = d >> 8;
    int nt = tile & (NT_N - 1), kt = tile / NT_N;
    int koff = (j < 4) ? (((l >> 4) << 3) + 2 * j)
                       : (16 + ((l >> 4) << 3) + 2 * (j - 4));
    k = kt * 32 + koff;
    n = (nt << 4) + (l & 15);
}

__global__ void k_pack_w1(const float* __restrict__ B_re,
                          const float* __restrict__ B_im,
                          const float* __restrict__ gamma_log,
                          uint32_t* __restrict__ wp) {
    int d = blockIdx.x * blockDim.x + threadIdx.x;
    if (d >= 32 * NT_N * 256) return;
    int k, n; unpack_idx(d, k, n);
    float g = __expf(gamma_log[n & (N_ST - 1)]);
    const float* src = (n < N_ST) ? B_re : B_im;
    int nn = n & (N_ST - 1);
    wp[d] = pack2bf(g * src[(size_t)nn * D_IN + k],
                    g * src[(size_t)nn * D_IN + k + 1]);
}

__device__ __forceinline__ float w2_fetch(const float* __restrict__ C_re,
                                          const float* __restrict__ C_im,
                                          const float* __restrict__ D,
                                          int n, int k) {
    if (k < N_ST)      return  C_re[(size_t)n * N_ST + k];
    if (k < 2 * N_ST)  return -C_im[(size_t)n * N_ST + (k - N_ST)];
    return D[(size_t)n * D_IN + (k - 2 * N_ST)];
}

__global__ void k_pack_w2(const float* __restrict__ C_re,
                          const float* __restrict__ C_im,
                          const float* __restrict__ D,
                          uint32_t* __restrict__ wp) {
    int d = blockIdx.x * blockDim.x + threadIdx.x;
    if (d >= 64 * NT_N * 256) return;
    int k, n; unpack_idx(d, k, n);
    wp[d] = pack2bf(w2_fetch(C_re, C_im, D, n, k),
                    w2_fetch(C_re, C_im, D, n, k + 1));
}

// ===========================================================================
// GEMM 1: Bu = xbf @ W1 -> bf16 planes p_re/p_im [M_TOT, N_ST]
// Block 512 thr = 16 waves; wave w -> 64-col group w; wave tile 64x64.
// A panel 64x1024 bf16 = 128 KB dynamic LDS, staged ONCE per 64-row block.
// ===========================================================================
__global__ __launch_bounds__(512, 1)
void k_gemm_in(const unsigned short* __restrict__ xbf,
               const uint32_t* __restrict__ wp,
               unsigned short* __restrict__ p_re,
               unsigned short* __restrict__ p_im) {
    extern __shared__ __align__(16) unsigned short panel[];     // 64 x 1024
    const int lane  = threadIdx.x & 31;
    const int ng    = threadIdx.x >> 5;           // wave id = col group 0..15
    const int mbase = blockIdx.x * MW;

    {   // flat copy: 64 rows x 1024 bf16 = 8192 uint4
        const uint4* src = (const uint4*)(xbf + (size_t)mbase * D_IN);
        uint4* dst = (uint4*)panel;
        for (int i = threadIdx.x; i < 8192; i += 512) dst[i] = src[i];
    }
    __syncthreads();

    v8f acc[4][4];
    #pragma unroll
    for (int mi = 0; mi < 4; ++mi)
        #pragma unroll
        for (int f = 0; f < 4; ++f) acc[mi][f] = v8f{};

    for (int kt = 0; kt < 32; ++kt) {
        v16bf a0 = lds_a_frag(panel, D_IN,  0, kt * 32, lane);
        v16bf a1 = lds_a_frag(panel, D_IN, 16, kt * 32, lane);
        v16bf a2 = lds_a_frag(panel, D_IN, 32, kt * 32, lane);
        v16bf a3 = lds_a_frag(panel, D_IN, 48, kt * 32, lane);
        #pragma unroll
        for (int f = 0; f < 4; ++f) {
            v16bf b = load_b_frag(wp, kt, ng * 4 + f, lane);
            acc[0][f] = wmma_bf16(a0, b, acc[0][f]);
            acc[1][f] = wmma_bf16(a1, b, acc[1][f]);
            acc[2][f] = wmma_bf16(a2, b, acc[2][f]);
            acc[3][f] = wmma_bf16(a3, b, acc[3][f]);
        }
    }
    #pragma unroll
    for (int f = 0; f < 4; ++f) {
        int n = ng * 64 + f * 16 + (lane & 15);
        unsigned short* outp = (n < N_ST) ? p_re : p_im;
        int nn = n & (N_ST - 1);
        #pragma unroll
        for (int mi = 0; mi < 4; ++mi)
            #pragma unroll
            for (int j = 0; j < 8; ++j) {
                int m = mbase + mi * 16 + j + 8 * (lane >> 4);
                outp[(size_t)m * N_ST + nn] = f2bf_bits(acc[mi][f][j]);
            }
    }
}

// ===========================================================================
// Scan pass 1: per (b, chunk, n) local scan (zero init) -> chunk finals
// ===========================================================================
__global__ void k_scan_finals(const unsigned short* __restrict__ bre,
                              const unsigned short* __restrict__ bim,
                              const float* __restrict__ lam_re,
                              const float* __restrict__ lam_im,
                              float* __restrict__ fre, float* __restrict__ fim) {
    int tid = blockIdx.x * blockDim.x + threadIdx.x;   // 262144
    int n = tid & (N_ST - 1);
    int c = (tid >> 9) & (NCHNK - 1);
    int b = tid >> 15;
    float lr = lam_re[n], li = lam_im[n];
    size_t base = ((size_t)(b * T_LEN + c * CHUNK)) * N_ST + n;
    float sr = 0.f, si = 0.f;
    for (int t = 0; t < CHUNK; ++t) {
        float br = bf2f(bre[base + (size_t)t * N_ST]);
        float bi = bf2f(bim[base + (size_t)t * N_ST]);
        float nr = fmaf(lr, sr, fmaf(-li, si, br));
        float ni = fmaf(lr, si, fmaf(li, sr, bi));
        sr = nr; si = ni;
    }
    size_t f = ((size_t)(b * NCHNK + c)) * N_ST + n;
    fre[f] = sr; fim[f] = si;
}

// ===========================================================================
// Scan pass 2: per (b, n) serial scan over chunk finals with lam^64
// ===========================================================================
__global__ void k_scan_carries(const float* __restrict__ fre,
                               const float* __restrict__ fim,
                               const float* __restrict__ lamL_re,
                               const float* __restrict__ lamL_im,
                               float* __restrict__ cre, float* __restrict__ cim) {
    int tid = blockIdx.x * blockDim.x + threadIdx.x;   // 4096
    if (tid >= B_SZ * N_ST) return;
    int n = tid & (N_ST - 1);
    int b = tid >> 9;
    float Lr = lamL_re[n], Li = lamL_im[n];
    float cr = 0.f, ci = 0.f;
    for (int c = 0; c < NCHNK; ++c) {
        size_t idx = ((size_t)(b * NCHNK + c)) * N_ST + n;
        cre[idx] = cr; cim[idx] = ci;
        float fr = fre[idx], fi = fim[idx];
        float nr = fmaf(Lr, cr, fmaf(-Li, ci, fr));
        float ni = fmaf(Lr, ci, fmaf(Li, cr, fi));
        cr = nr; ci = ni;
    }
}

// ===========================================================================
// Scan pass 3: rescan chunk with true carry, overwrite planes with states
// ===========================================================================
__global__ void k_scan_fixup(unsigned short* __restrict__ bre,
                             unsigned short* __restrict__ bim,
                             const float* __restrict__ lam_re,
                             const float* __restrict__ lam_im,
                             const float* __restrict__ cre,
                             const float* __restrict__ cim) {
    int tid = blockIdx.x * blockDim.x + threadIdx.x;   // 262144
    int n = tid & (N_ST - 1);
    int c = (tid >> 9) & (NCHNK - 1);
    int b = tid >> 15;
    float lr = lam_re[n], li = lam_im[n];
    size_t fidx = ((size_t)(b * NCHNK + c)) * N_ST + n;
    float sr = cre[fidx], si = cim[fidx];
    size_t base = ((size_t)(b * T_LEN + c * CHUNK)) * N_ST + n;
    for (int t = 0; t < CHUNK; ++t) {
        size_t a = base + (size_t)t * N_ST;
        float br = bf2f(bre[a]), bi = bf2f(bim[a]);
        float nr = fmaf(lr, sr, fmaf(-li, si, br));
        float ni = fmaf(lr, si, fmaf(li, sr, bi));
        sr = nr; si = ni;
        bre[a] = f2bf_bits(sr); bim[a] = f2bf_bits(si);
    }
}

// ===========================================================================
// GEMM 2: y = [st_re | st_im | xbf] @ W2  (K=2048)
// Block 512 thr = 16 waves; wave tile 64x64; A panel 64x2048 = 256 KB LDS.
// ===========================================================================
__global__ __launch_bounds__(512, 1)
void k_gemm_out(const unsigned short* __restrict__ sre,
                const unsigned short* __restrict__ sim,
                const unsigned short* __restrict__ xbf,
                const uint32_t* __restrict__ wp,
                float* __restrict__ y) {
    extern __shared__ __align__(16) unsigned short panel[];     // 64 x 2048
    const int lane  = threadIdx.x & 31;
    const int ng    = threadIdx.x >> 5;           // wave id = col group 0..15
    const int mbase = blockIdx.x * MW;

    {   // row layout [m][k]: k 0..511 re | 512..1023 im | 1024..2047 x
        uint4* dst = (uint4*)panel;                     // row = 256 uint4
        const uint4* s0 = (const uint4*)(sre + (size_t)mbase * N_ST);
        const uint4* s1 = (const uint4*)(sim + (size_t)mbase * N_ST);
        const uint4* s2 = (const uint4*)(xbf + (size_t)mbase * D_IN);
        for (int i = threadIdx.x; i < 4096; i += 512) { // 64 rows x 64 uint4
            int m = i >> 6, c = i & 63;
            dst[m * 256 + c]      = s0[i];
            dst[m * 256 + 64 + c] = s1[i];
        }
        for (int i = threadIdx.x; i < 8192; i += 512) { // 64 rows x 128 uint4
            int m = i >> 7, c = i & 127;
            dst[m * 256 + 128 + c] = s2[i];
        }
    }
    __syncthreads();

    v8f acc[4][4];
    #pragma unroll
    for (int mi = 0; mi < 4; ++mi)
        #pragma unroll
        for (int f = 0; f < 4; ++f) acc[mi][f] = v8f{};

    for (int kt = 0; kt < 64; ++kt) {
        v16bf a0 = lds_a_frag(panel, 2048,  0, kt * 32, lane);
        v16bf a1 = lds_a_frag(panel, 2048, 16, kt * 32, lane);
        v16bf a2 = lds_a_frag(panel, 2048, 32, kt * 32, lane);
        v16bf a3 = lds_a_frag(panel, 2048, 48, kt * 32, lane);
        #pragma unroll
        for (int f = 0; f < 4; ++f) {
            v16bf b = load_b_frag(wp, kt, ng * 4 + f, lane);
            acc[0][f] = wmma_bf16(a0, b, acc[0][f]);
            acc[1][f] = wmma_bf16(a1, b, acc[1][f]);
            acc[2][f] = wmma_bf16(a2, b, acc[2][f]);
            acc[3][f] = wmma_bf16(a3, b, acc[3][f]);
        }
    }
    #pragma unroll
    for (int f = 0; f < 4; ++f) {
        int n = ng * 64 + f * 16 + (lane & 15);
        #pragma unroll
        for (int mi = 0; mi < 4; ++mi)
            #pragma unroll
            for (int j = 0; j < 8; ++j) {
                int m = mbase + mi * 16 + j + 8 * (lane >> 4);
                y[(size_t)m * D_OUT + n] = acc[mi][f][j];
            }
    }
}

// ===========================================================================
// Host-side launcher
// ===========================================================================
extern "C" void kernel_launch(void* const* d_in, const int* in_sizes, int n_in,
                              void* d_out, int out_size, void* d_ws, size_t ws_size,
                              hipStream_t stream) {
    const float* x         = (const float*)d_in[0];
    const float* nu_log    = (const float*)d_in[1];
    const float* theta_log = (const float*)d_in[2];
    const float* gamma_log = (const float*)d_in[3];
    const float* B_re      = (const float*)d_in[4];
    const float* B_im      = (const float*)d_in[5];
    const float* C_re      = (const float*)d_in[6];
    const float* C_im      = (const float*)d_in[7];
    const float* Dm        = (const float*)d_in[8];
    float* y = (float*)d_out;

    // workspace layout
    const size_t S  = (size_t)M_TOT * N_ST;              // 16,777,216 states
    const size_t F  = (size_t)B_SZ * NCHNK * N_ST;       // 262,144
    unsigned short* p_re = (unsigned short*)d_ws;        // bf16 Bu/state planes
    unsigned short* p_im = p_re + S;
    unsigned short* xbf  = p_im + S;                     // bf16 x [M_TOT, D_IN]
    float* fre     = (float*)(xbf + (size_t)M_TOT * D_IN);
    float* fim     = fre + F;
    float* cre     = fim + F;
    float* cim     = cre + F;
    float* lam_re  = cim + F;
    float* lam_im  = lam_re + N_ST;
    float* lamL_re = lam_im + N_ST;
    float* lamL_im = lamL_re + N_ST;
    uint32_t* wp1  = (uint32_t*)(lamL_im + N_ST);        // 524,288 dwords
    uint32_t* wp2  = wp1 + (size_t)32 * NT_N * 256;      // 1,048,576 dwords

    // 1) streaming conversion / small prep
    k_cvt_x<<<((size_t)M_TOT * D_IN / 8) / 256, 256, 0, stream>>>(x, xbf);
    k_lam<<<2, 256, 0, stream>>>(nu_log, theta_log, lam_re, lam_im, lamL_re, lamL_im);
    k_pack_w1<<<(32 * NT_N * 256) / 256, 256, 0, stream>>>(B_re, B_im, gamma_log, wp1);
    k_pack_w2<<<(64 * NT_N * 256) / 256, 256, 0, stream>>>(C_re, C_im, Dm, wp2);
    // 2) input GEMM -> Bu planes (bf16); 128 KB dynamic LDS, 16 waves/block
    k_gemm_in<<<M_TOT / MW, 512, 64 * D_IN * 2, stream>>>(xbf, wp1, p_re, p_im);
    // 3) chunked scan over T
    k_scan_finals<<<(B_SZ * NCHNK * N_ST) / 256, 256, 0, stream>>>(
        p_re, p_im, lam_re, lam_im, fre, fim);
    k_scan_carries<<<(B_SZ * N_ST) / 256, 256, 0, stream>>>(
        fre, fim, lamL_re, lamL_im, cre, cim);
    k_scan_fixup<<<(B_SZ * NCHNK * N_ST) / 256, 256, 0, stream>>>(
        p_re, p_im, lam_re, lam_im, cre, cim);
    // 4) output GEMM -> y; 256 KB dynamic LDS, 16 waves/block
    k_gemm_out<<<M_TOT / MW, 512, 64 * 2048 * 2, stream>>>(p_re, p_im, xbf, wp2, y);
    (void)in_sizes; (void)n_in; (void)out_size; (void)ws_size;
}
